// MHConvAttention_738734375506
// MI455X (gfx1250) — matrix operand order repaired and checked by reading, not compiled, not run on hardware
//
#include <hip/hip_runtime.h>

// ---------------------------------------------------------------------------
// MI455X (gfx1250) lambda-attention block: bf16 WMMA pipeline with
// double-buffered LDS, vectorized staging, and async global->LDS B-tiles.
// ---------------------------------------------------------------------------

typedef __bf16 bf16;
typedef __attribute__((ext_vector_type(16))) __bf16 v16bf;
typedef __attribute__((ext_vector_type(8)))  __bf16 v8bf;
typedef __attribute__((ext_vector_type(8)))  float  v8f;

namespace {
constexpr int kB   = 16;
constexpr int kC   = 256;
constexpr int kH   = 64;
constexpr int kW   = 64;
constexpr int kHW  = kH * kW;      // 4096
constexpr int kNH  = 8;
constexpr int kHD  = kC / kNH;     // 32
constexpr int kBP  = kB * kNH;     // 128 heads total
constexpr float kScale = 0.17677669529663687f; // 32^-0.5
}

enum { GEMM_CPE = 0, GEMM_QKV = 1, GEMM_OUT = 2 };

// ------------------ async global->LDS (CDNA5, ASYNCcnt) --------------------
__device__ __forceinline__ void async_g2l_b128(unsigned lds_off, const void* g) {
  asm volatile("global_load_async_to_lds_b128 %0, %1, off"
               :: "v"(lds_off), "v"((unsigned long long)(size_t)g) : "memory");
}
__device__ __forceinline__ void wait_async0() {
  asm volatile("s_wait_asynccnt 0x0" ::: "memory");
}

// --------------------------- fp32 -> bf16 convert ---------------------------
__global__ __launch_bounds__(256)
void cvt_f32_bf16_v8(const float* __restrict__ in, bf16* __restrict__ out, int n) {
  int i = (blockIdx.x * 256 + threadIdx.x) * 8;
  if (i < n) {
    v8f f; __builtin_memcpy(&f, in + i, 32);
    v8bf h;
    #pragma unroll
    for (int r = 0; r < 8; ++r) h[r] = (bf16)f[r];
    __builtin_memcpy(out + i, &h, 16);
  }
}

// cpe_w (O,C,3,3) fp32 -> bf16 transposed to [tap][o][c] for contiguous B tiles
__global__ __launch_bounds__(256)
void cvt_cpe_w(const float* __restrict__ in, bf16* __restrict__ out) {
  int idx = blockIdx.x * 256 + threadIdx.x;         // tap*65536 + o*256 + c
  if (idx < kC * kC * 9) {
    int tap = idx >> 16;
    int oc  = idx & 0xFFFF;                         // o*256 + c
    out[idx] = (bf16)in[(size_t)oc * 9 + tap];
  }
}

// ------------------------------- fused GEMM --------------------------------
// 256 threads (8 wave32); output tile 128(M=spatial) x 64(N=channels);
// wave = 32x32 subtile (2x2 v_wmma_f32_16x16x32_bf16). Double-buffered LDS,
// A staged via registers (b128 global loads), B via global_load_async_to_lds.
template <int MODE>
__global__ __launch_bounds__(256)
void gemm_wmma(const bf16*  __restrict__ Act,   // [B][C][HW] bf16
               const bf16*  __restrict__ Wt,    // bf16 weights (CPE: [tap][o][c])
               const float* __restrict__ bias,
               const float* __restrict__ resid, // fp32 src (CPE residual)
               bf16*  __restrict__ Ob,
               bf16*  __restrict__ Oq, bf16* __restrict__ Ok, bf16* __restrict__ Ov,
               float* __restrict__ Of) {
  __shared__ bf16 As[2][128][40];   // 80B row stride keeps 16B alignment
  __shared__ bf16 Bs[2][64][40];

  const int t     = threadIdx.x;
  const int b     = blockIdx.z;
  const int mBase = blockIdx.x * 128;
  const int nBase = blockIdx.y * 64;

  const int wave  = t >> 5, lane = t & 31;
  const int mi    = wave >> 1, ni = wave & 1;
  const int lhalf = lane >> 4, llow = lane & 15;

  // A staging: 2 chunks of (8 consecutive m, one k) per thread
  const int mA  = (t & 15) * 8;
  const int kA  = t >> 4;                 // 0..15 ; chunks use kA and kA+16
  // B staging: one 16B chunk (8 consecutive k, one n) per thread
  const int nB  = t >> 2;
  const int kB0 = (t & 3) * 8;

  bf16 va[2][8];

  auto stageA = [&](int step) {
    const int tap = (MODE == GEMM_CPE) ? (step >> 3) : 0;
    const int kc  = (MODE == GEMM_CPE) ? (step & 7) : step;
    const int dy  = tap / 3 - 1, dx = tap % 3 - 1;
    #pragma unroll
    for (int u = 0; u < 2; ++u) {
      const int c = kc * 32 + kA + u * 16;
      const bf16* base = Act + ((size_t)b * kC + c) * kHW;
      if (MODE == GEMM_CPE) {
        const int p0 = mBase + mA;
        const int h = p0 >> 6, w0 = p0 & 63;
        const int hh = h + dy;
        const bool rowok = (hh >= 0) && (hh < kH);
        const bool inner = rowok && !((dx < 0) && (w0 == 0))
                                 && !((dx > 0) && (w0 == 56));
        if (inner) {
          __builtin_memcpy(&va[u][0], base + (hh << 6) + w0 + dx, 16);
        } else {
          #pragma unroll
          for (int e = 0; e < 8; ++e) {
            int ww = w0 + e + dx;
            va[u][e] = (rowok && ww >= 0 && ww < kW) ? base[(hh << 6) + ww]
                                                     : (bf16)0.0f;
          }
        }
      } else {
        __builtin_memcpy(&va[u][0], base + mBase + mA, 16);
      }
    }
  };

  auto scatterA = [&](int buf) {
    #pragma unroll
    for (int u = 0; u < 2; ++u) {
      const int k = kA + u * 16;
      #pragma unroll
      for (int e = 0; e < 8; ++e) As[buf][mA + e][k] = va[u][e];
    }
  };

  auto asyncB = [&](int step, int buf) {
    const int tap = (MODE == GEMM_CPE) ? (step >> 3) : 0;
    const int kc  = (MODE == GEMM_CPE) ? (step & 7) : step;
    const bf16* g = (MODE == GEMM_CPE)
        ? Wt + (size_t)tap * kC * kC + (size_t)(nBase + nB) * kC + kc * 32 + kB0
        : Wt + (size_t)(nBase + nB) * kC + kc * 32 + kB0;
    async_g2l_b128((unsigned)(size_t)&Bs[buf][nB][kB0], g);
  };

  v8f acc[2][2] = {};
  auto compute = [&](int buf) {
    #pragma unroll
    for (int im = 0; im < 2; ++im) {
      const int row = mi * 32 + im * 16 + llow;
      v16bf a;
      __builtin_memcpy(&a, &As[buf][row][lhalf * 16], 16);
      __builtin_memcpy((char*)&a + 16, &As[buf][row][lhalf * 16 + 8], 16);
      #pragma unroll
      for (int in = 0; in < 2; ++in) {
        const int col = ni * 32 + in * 16 + llow;
        v16bf bb;
        __builtin_memcpy(&bb, &Bs[buf][col][lhalf * 16], 32);
        acc[im][in] = __builtin_amdgcn_wmma_f32_16x16x32_bf16(
            false, a, false, bb, (short)0, acc[im][in], false, false);
      }
    }
  };

  const int nsteps = (MODE == GEMM_CPE) ? 72 : 8;
  stageA(0);
  asyncB(0, 0);
  int buf = 0;
  for (int step = 0; step < nsteps; ++step) {
    scatterA(buf);
    wait_async0();            // this wave's async B-tile has landed in LDS
    __syncthreads();          // all waves' A scatters + B tiles visible
    if (step + 1 < nsteps) {  // overlap next-step loads with the WMMAs
      asyncB(step + 1, buf ^ 1);
      stageA(step + 1);
    }
    compute(buf);
    buf ^= 1;
  }

  // Epilogue: lane's 8 acc values sit at 8 consecutive p for one channel o.
  #pragma unroll
  for (int im = 0; im < 2; ++im) {
    const int pb = mBase + mi * 32 + im * 16 + lhalf * 8;
    #pragma unroll
    for (int in = 0; in < 2; ++in) {
      const int o = nBase + ni * 32 + in * 16 + llow;
      v8f v = acc[im][in];
      if (MODE == GEMM_CPE) {
        const size_t off = ((size_t)b * kC + o) * kHW + pb;
        v8f rv; __builtin_memcpy(&rv, resid + off, 32);
        const float bo = bias[o];
        v8bf pk;
        #pragma unroll
        for (int r = 0; r < 8; ++r) pk[r] = (bf16)(v[r] + bo + rv[r]);
        __builtin_memcpy(Ob + off, &pk, 16);
      } else if (MODE == GEMM_QKV) {
        const size_t off = ((size_t)b * kC + (o & 255)) * kHW + pb;
        v8bf pk;
        #pragma unroll
        for (int r = 0; r < 8; ++r) pk[r] = (bf16)v[r];
        bf16* dst = (o < 256) ? Oq : (o < 512) ? Ok : Ov;
        __builtin_memcpy(dst + off, &pk, 16);
      } else {
        const size_t off = ((size_t)b * kC + o) * kHW + pb;
        const float bo = bias[o];
        #pragma unroll
        for (int r = 0; r < 8; ++r) v[r] += bo;
        __builtin_memcpy(Of + off, &v, 32);
      }
    }
  }
}

// --------------------- softmax over spatial (in-place) ---------------------
__global__ __launch_bounds__(256)
void softmax_rows(bf16* __restrict__ k) {       // B*C rows of length HW
  __shared__ float red[256];
  const int t = threadIdx.x;
  bf16* kr = k + (size_t)blockIdx.x * kHW + t * 16;

  v16bf x; __builtin_memcpy(&x, kr, 32);        // contiguous 32B per thread
  float mx = -3.4e38f;
  #pragma unroll
  for (int j = 0; j < 16; ++j) mx = fmaxf(mx, (float)x[j]);
  red[t] = mx; __syncthreads();
  for (int s = 128; s > 0; s >>= 1) {
    if (t < s) red[t] = fmaxf(red[t], red[t + s]);
    __syncthreads();
  }
  mx = red[0]; __syncthreads();

  float ex[16], sum = 0.f;
  #pragma unroll
  for (int j = 0; j < 16; ++j) { ex[j] = __expf((float)x[j] - mx); sum += ex[j]; }
  red[t] = sum; __syncthreads();
  for (int s = 128; s > 0; s >>= 1) {
    if (t < s) red[t] += red[t + s];
    __syncthreads();
  }
  const float inv = 1.f / red[0];

  v16bf y;
  #pragma unroll
  for (int j = 0; j < 16; ++j) y[j] = (bf16)(ex[j] * inv);
  __builtin_memcpy(kr, &y, 32);
}

// --------- content lambda: lamT[b'][o][i] = scale * sum_n kf*v -------------
// Per head: M=32(i), N=32(o), K=4096 — both operands K-contiguous in global.
__global__ __launch_bounds__(128)
void lambda_wmma(const bf16* __restrict__ kf, const bf16* __restrict__ v,
                 bf16* __restrict__ lamT, float scale) {
  const int bp   = blockIdx.x;
  const int t    = threadIdx.x;
  const int wave = t >> 5, lane = t & 31;
  const int mi = wave >> 1, ni = wave & 1;
  const int lhalf = lane >> 4, llow = lane & 15;

  const bf16* kfb = kf + (size_t)bp * kHD * kHW;
  const bf16* vb  = v  + (size_t)bp * kHD * kHW;
  const int rowi = mi * 16 + llow;
  const int colo = ni * 16 + llow;
  const int k0 = lhalf * 16, ks = lhalf * 16;

  v8f acc = {};
  #pragma unroll 4
  for (int kk = 0; kk < kHW; kk += 32) {
    v16bf a, bb;
    const bf16* pa = kfb + (size_t)rowi * kHW + kk + k0;
    __builtin_memcpy(&a, pa, 16);
    __builtin_memcpy((char*)&a + 16, pa + 8, 16);
    __builtin_memcpy(&bb, vb + (size_t)colo * kHW + kk + ks, 32);
    acc = __builtin_amdgcn_wmma_f32_16x16x32_bf16(
        false, a, false, bb, (short)0, acc, false, false);
  }
  #pragma unroll
  for (int r = 0; r < 8; ++r) {
    int i = mi * 16 + r + lhalf * 8;
    int o = ni * 16 + llow;
    lamT[(size_t)bp * kHD * kHD + o * kHD + i] = (bf16)(acc[r] * scale);
  }
}

// -------------------- depthwise 5x5 positional lambda ----------------------
// Block covers 4 image rows (256 px); 8-row window staged in LDS.
__global__ __launch_bounds__(256)
void dwconv5(const bf16* __restrict__ v, const float* __restrict__ rel,
             bf16* __restrict__ pos) {
  __shared__ float rw[25];
  __shared__ bf16 tile[8][64];
  const int b = blockIdx.z, c = blockIdx.y;
  const int t = threadIdx.x;
  const int h0 = blockIdx.x * 4;                 // first output row
  const bf16* vb = v + ((size_t)b * kC + c) * kHW;

  if (t < 25) rw[t] = rel[(c & (kHD - 1)) * 25 + t];
  #pragma unroll
  for (int j = 0; j < 2; ++j) {
    int idx = t + j * 256;                       // 0..511
    int r = idx >> 6, w = idx & 63;
    int hh = h0 - 2 + r;
    tile[r][w] = (hh >= 0 && hh < kH) ? vb[(hh << 6) + w] : (bf16)0.0f;
  }
  __syncthreads();

  const int hr = t >> 6;                         // 0..3 output row in block
  const int w  = t & 63;
  float acc = 0.f;
  #pragma unroll
  for (int dy = 0; dy < 5; ++dy) {
    #pragma unroll
    for (int dx = 0; dx < 5; ++dx) {
      int ww = w + dx - 2;
      if (ww >= 0 && ww < kW)
        acc += rw[dy * 5 + dx] * (float)tile[hr + dy][ww];
    }
  }
  pos[((size_t)b * kC + c) * kHW + ((h0 + hr) << 6) + w] = (bf16)acc;
}

// --- content apply + positional term: res = q . lamT + q * pos_lambda ------
// Per head: M=4096(p), N=32(o), K=32(i) -> one WMMA K-step per tile.
__global__ __launch_bounds__(256)
void content_wmma(const bf16* __restrict__ q, const bf16* __restrict__ lamT,
                  const bf16* __restrict__ pos, bf16* __restrict__ res) {
  __shared__ bf16 As[64][40];                    // q^T tile: [p][i]
  const int bp    = blockIdx.y;
  const int pBase = blockIdx.x * 64;
  const int t = threadIdx.x;

  const bf16* qb = q + (size_t)bp * kHD * kHW;
  {                                              // one 16B chunk per thread
    const int m0 = (t & 7) * 8, i = t >> 3;
    bf16 va[8];
    __builtin_memcpy(va, qb + (size_t)i * kHW + pBase + m0, 16);
    #pragma unroll
    for (int e = 0; e < 8; ++e) As[m0 + e][i] = va[e];
  }
  __syncthreads();

  const int wave = t >> 5, lane = t & 31;
  const int mi = wave >> 1, ni = wave & 1;
  const int lhalf = lane >> 4, llow = lane & 15;

  v16bf a;
  const int row = mi * 16 + llow;
  __builtin_memcpy(&a, &As[row][lhalf * 16], 16);
  __builtin_memcpy((char*)&a + 16, &As[row][lhalf * 16 + 8], 16);

  const int n = ni * 16 + llow;
  v16bf bb;
  __builtin_memcpy(&bb, lamT + (size_t)bp * kHD * kHD + n * kHD + lhalf * 16, 32);

  v8f acc = {};
  acc = __builtin_amdgcn_wmma_f32_16x16x32_bf16(
      false, a, false, bb, (short)0, acc, false, false);

  const bf16* posb = pos + (size_t)bp * kHD * kHW;
  bf16*       resb = res + (size_t)bp * kHD * kHW;
  const int pb = pBase + mi * 16 + lhalf * 8;
  v8bf pv; __builtin_memcpy(&pv, posb + (size_t)n * kHW + pb, 16);
  v8bf outv;
  #pragma unroll
  for (int r = 0; r < 8; ++r) {
    float qv = (float)As[mi * 16 + lhalf * 8 + r][n];
    outv[r] = (bf16)(acc[r] + qv * (float)pv[r]);
  }
  __builtin_memcpy(resb + (size_t)n * kHW + pb, &outv, 16);
}

// ---------------------------------------------------------------------------
extern "C" void kernel_launch(void* const* d_in, const int* in_sizes, int n_in,
                              void* d_out, int out_size, void* d_ws, size_t ws_size,
                              hipStream_t stream) {
  (void)in_sizes; (void)n_in; (void)out_size; (void)ws_size;
  const float* src   = (const float*)d_in[0];
  const float* rel   = (const float*)d_in[1];
  const float* qkv_w = (const float*)d_in[2];
  const float* cpe_w = (const float*)d_in[3];
  const float* cpe_b = (const float*)d_in[4];
  const float* out_w = (const float*)d_in[5];
  const float* out_b = (const float*)d_in[6];
  float* out = (float*)d_out;

  const size_t NACT = (size_t)kB * kC * kHW;   // 16,777,216 elems
  const size_t ABYT = NACT * sizeof(bf16);     // 32 MB per activation buffer

  char* ws = (char*)d_ws;
  bf16* srcb = (bf16*)(ws + 0 * ABYT);         // src bf16; reused as pos_lambda
  bf16* xb   = (bf16*)(ws + 1 * ABYT);         // cpe output; reused as result
  bf16* qb   = (bf16*)(ws + 2 * ABYT);
  bf16* kb   = (bf16*)(ws + 3 * ABYT);         // softmax in place
  bf16* vb   = (bf16*)(ws + 4 * ABYT);
  char* wsp  = ws + 5 * ABYT;
  bf16* wqb  = (bf16*)wsp;              wsp += (size_t)768 * 256 * 2;
  bf16* wcb  = (bf16*)wsp;              wsp += (size_t)256 * 256 * 9 * 2;
  bf16* wob  = (bf16*)wsp;              wsp += (size_t)256 * 256 * 2;
  bf16* lamT = (bf16*)wsp;                     // 128*32*32 bf16

  // 1) converts (cpe weights transposed to [tap][o][c])
  cvt_f32_bf16_v8<<<(int)(NACT / 2048), 256, 0, stream>>>(src, srcb, (int)NACT);
  cvt_f32_bf16_v8<<<768 * 256 / 2048, 256, 0, stream>>>(qkv_w, wqb, 768 * 256);
  cvt_f32_bf16_v8<<<256 * 256 / 2048, 256, 0, stream>>>(out_w, wob, 256 * 256);
  cvt_cpe_w<<<(256 * 256 * 9 + 255) / 256, 256, 0, stream>>>(cpe_w, wcb);

  // 2) cpe 3x3 implicit GEMM + bias + residual -> x
  gemm_wmma<GEMM_CPE><<<dim3(kHW / 128, 256 / 64, kB), 256, 0, stream>>>(
      srcb, wcb, cpe_b, src, xb, nullptr, nullptr, nullptr, nullptr);

  // 3) qkv projection (N=768), split store
  gemm_wmma<GEMM_QKV><<<dim3(kHW / 128, 768 / 64, kB), 256, 0, stream>>>(
      xb, wqb, nullptr, nullptr, nullptr, qb, kb, vb, nullptr);

  // 4) softmax over spatial positions of k (in place)
  softmax_rows<<<kB * kC, 256, 0, stream>>>(kb);

  // 5) content lambda per head (scale folded, stored transposed)
  lambda_wmma<<<kBP, 128, 0, stream>>>(kb, vb, lamT, kScale);

  // 6) depthwise 5x5 positional lambda on v (reuses srcb)
  dwconv5<<<dim3(kH / 4, kC, kB), 256, 0, stream>>>(vb, rel, srcb);

  // 7) content apply + q*pos_lambda -> result (reuses xb)
  content_wmma<<<dim3(kHW / 64, kBP), 256, 0, stream>>>(qb, lamT, srcb, xb);

  // 8) output projection + bias -> fp32 out
  gemm_wmma<GEMM_OUT><<<dim3(kHW / 128, 256 / 64, kB), 256, 0, stream>>>(
      xb, wob, out_b, nullptr, nullptr, nullptr, nullptr, nullptr, out);
}
